// SelfAttention_87875030876732
// MI455X (gfx1250) — matrix operand order repaired and checked by reading, not compiled, not run on hardware
//
#include <hip/hip_runtime.h>
#include <hip/hip_bf16.h>

typedef __bf16 bf16;
typedef __attribute__((ext_vector_type(16))) __bf16 bf16x16;
typedef __attribute__((ext_vector_type(8)))  __bf16 bf16x8;
typedef __attribute__((ext_vector_type(8)))  float  f32x8;
typedef __attribute__((ext_vector_type(4)))  unsigned int u32x4;
typedef __attribute__((ext_vector_type(8)))  int i32x8;
typedef __attribute__((ext_vector_type(4)))  int i32x4;

#define NB 4
#define NS 4096
#define ND 512
#define NKB (NS / 64)

#if __has_builtin(__builtin_amdgcn_tensor_load_to_lds) && __has_builtin(__builtin_amdgcn_s_wait_tensorcnt)
#define HAVE_TDM 1
#else
#define HAVE_TDM 0
#endif

// LDS layout (bytes) for flash_attn
#define KS_BYTES   65536            // one K buffer: 64x512 bf16
#define VT_OFF     (2 * KS_BYTES)   // 131072 : Vt [512][72] bf16
#define QS_OFF     (VT_OFF + 512*72*2)        // 204800 : Qs 64x512 bf16
#define PS_OFF     (QS_OFF + KS_BYTES)        // 270336 : P staging 8 x 16*64 bf16
#define SMEM_TOTAL (PS_OFF + 8*16*64*2)       // 286720

__device__ __forceinline__ bf16x16 ldpair(const bf16* p0, const bf16* p1) {
  bf16x8 lo = *(const bf16x8*)p0;
  bf16x8 hi = *(const bf16x8*)p1;
  return __builtin_shufflevector(lo, hi, 0,1,2,3,4,5,6,7,8,9,10,11,12,13,14,15);
}

#if HAVE_TDM
// 1-D contiguous TDM copy: global -> LDS, nbytes must be a multiple of 8 and
// nbytes/8 <= 65535 (tile_dim0 is 16-bit).  D# per CDNA5 ISA.
__device__ __forceinline__ void tdm_load_1d(unsigned lds_off, const void* gaddr, unsigned nbytes) {
  unsigned long long ga = (unsigned long long)gaddr;
  unsigned n8 = nbytes >> 3;                       // elements of 8 bytes
  u32x4 g0 = { 1u,                                 // count=1 (valid user D#)
               lds_off,                            // lds_addr [63:32]
               (unsigned)(ga & 0xFFFFFFFFu),       // global_addr lo
               (unsigned)((ga >> 32) & 0x1FFFFFFu) | (2u << 30) }; // addr hi | type=2
  i32x8 g1 = { (int)(3u << 16),                    // wg_mask=0, data_size=3 (8B)
               (int)((n8 & 0xFFFFu) << 16),        // tensor_dim0 [15:0] at bits[63:48]
               (int)(((n8 >> 16) & 0xFFFFu) | (1u << 16)), // tensor_dim0 hi | tensor_dim1=1
               (int)((n8 & 0xFFFFu) << 16),        // tensor_dim1 hi(0) | tile_dim0
               0, 0, 0, 0 };                       // tile_dim1/2=0, strides unused
  i32x4 z4 = {0,0,0,0};
#if __clang_major__ >= 23
  i32x8 z8 = {0,0,0,0,0,0,0,0};
  __builtin_amdgcn_tensor_load_to_lds(g0, g1, z4, z4, z8, 0);
#else
  __builtin_amdgcn_tensor_load_to_lds(g0, g1, z4, z4, 0);
#endif
}
#endif

__global__ void cvt_f32_bf16x8(const f32x8* __restrict__ src, bf16x8* __restrict__ dst, int n8) {
  int i = blockIdx.x * blockDim.x + threadIdx.x;
  if (i < n8) dst[i] = __builtin_convertvector(src[i], bf16x8);
}

// C = relu(x @ W^T + b) in bf16, software-pipelined one K-step ahead.
__global__ void __launch_bounds__(256)
qkv_gemm(const bf16* __restrict__ x,
         const bf16* __restrict__ w0, const bf16* __restrict__ w1, const bf16* __restrict__ w2,
         const float* __restrict__ b0, const float* __restrict__ b1, const float* __restrict__ b2,
         bf16* __restrict__ o0, bf16* __restrict__ o1, bf16* __restrict__ o2) {
  const bf16* W; const float* bias; bf16* O;
  if (blockIdx.z == 0)      { W = w0; bias = b0; O = o0; }
  else if (blockIdx.z == 1) { W = w1; bias = b1; O = o1; }
  else                      { W = w2; bias = b2; O = o2; }

  const int tid  = threadIdx.x;
  const int lane = tid & 31;
  const int wave = tid >> 5;
  const int ln16 = lane & 15;
  const int half = lane >> 4;
  const int m0   = blockIdx.x * 128 + wave * 16;
  const int n0   = blockIdx.y * 64;
  const int offA = half * 16;
  const int offB = half * 16;

  f32x8 c4[4];
  const f32x8 fz = {0.f,0.f,0.f,0.f,0.f,0.f,0.f,0.f};
  #pragma unroll
  for (int t = 0; t < 4; ++t) c4[t] = fz;

  const bf16* xr = x + (size_t)(m0 + ln16) * ND + offA;
  const bf16* wr[4];
  #pragma unroll
  for (int t = 0; t < 4; ++t) wr[t] = W + (size_t)(n0 + t*16 + ln16) * ND + offB;

  bf16x16 a = ldpair(xr, xr + 8);
  bf16x16 bm[4];
  #pragma unroll
  for (int t = 0; t < 4; ++t) bm[t] = ldpair(wr[t], wr[t] + 8);

  #pragma unroll
  for (int d0 = 0; d0 < ND; d0 += 32) {
    bf16x16 an = a;
    bf16x16 bn[4] = { bm[0], bm[1], bm[2], bm[3] };
    if (d0 + 32 < ND) {               // prefetch next K-step
      an = ldpair(xr + d0 + 32, xr + d0 + 40);
      #pragma unroll
      for (int t = 0; t < 4; ++t) bn[t] = ldpair(wr[t] + d0 + 32, wr[t] + d0 + 40);
    }
    #pragma unroll
    for (int t = 0; t < 4; ++t)
      c4[t] = __builtin_amdgcn_wmma_f32_16x16x32_bf16(
                  false, a, false, bm[t], (short)0, c4[t], false, false);
    a = an;
    #pragma unroll
    for (int t = 0; t < 4; ++t) bm[t] = bn[t];
  }

  #pragma unroll
  for (int t = 0; t < 4; ++t) {
    int e = n0 + t*16 + ln16;
    float bb = bias[e];
    #pragma unroll
    for (int r = 0; r < 8; ++r) {
      int row = m0 + r + 8*half;
      float vr = fmaxf(c4[t][r] + bb, 0.f);
      O[(size_t)row * ND + e] = (bf16)vr;
    }
  }
}

// Flash attention: block = 256 threads (8 waves), 64 queries per block.
// K tiles double-buffered via TDM prefetch; V transposed through VGPRs.
__global__ void __launch_bounds__(256)
flash_attn(const bf16* __restrict__ q, const bf16* __restrict__ k,
           const bf16* __restrict__ v, float* __restrict__ out) {
  extern __shared__ char smem_raw[];
  bf16* Vt = (bf16*)(smem_raw + VT_OFF);   // [512][72]
  bf16* Qs = (bf16*)(smem_raw + QS_OFF);   // [64][512]
  bf16* Ps = (bf16*)(smem_raw + PS_OFF);   // [8][16*64]

  const int tid   = threadIdx.x;
  const int lane  = tid & 31;
  const int wave  = tid >> 5;
  const int ln16  = lane & 15;
  const int half  = lane >> 4;
  const int b     = blockIdx.y;
  const int q0g   = blockIdx.x * 64;
  const int qsub  = wave >> 1;
  const int chalf = wave & 1;
  const size_t bbase = (size_t)b * NS * ND;
  const int offA = half * 16;
  const int offB = half * 16;

#if HAVE_TDM
  if (wave == 0) tdm_load_1d(0, k + bbase, KS_BYTES);                 // K tile 0
  if (wave == 1) {
    tdm_load_1d(QS_OFF, q + bbase + (size_t)q0g * ND, KS_BYTES);      // Q tile
    __builtin_amdgcn_s_wait_tensorcnt(0);
  }
#else
  for (int i = 0; i < 16; ++i) {
    int cc = tid + i*256;
    *(bf16x8*)(Qs + cc*8) = *(const bf16x8*)(q + bbase + (size_t)q0g*ND + cc*8);
  }
#endif

  const f32x8 fz = {0.f,0.f,0.f,0.f,0.f,0.f,0.f,0.f};
  f32x8 acc[16];
  #pragma unroll
  for (int c = 0; c < 16; ++c) acc[c] = fz;
  float m_[8], l_[8];
  #pragma unroll
  for (int r = 0; r < 8; ++r) { m_[r] = -3.0e38f; l_[r] = 0.f; }

  const bf16* QsRow = Qs + (size_t)(qsub*16 + ln16) * ND + offA;
  bf16* Pw = Ps + wave * (16*64);
  const float sc2 = 0.044194173824159216f * 1.4426950408889634f; // 1/sqrt(512)*log2e

  for (int kb = 0; kb < NKB; ++kb) {
    __syncthreads();   // previous compute done; next K buffer + Vt reusable
#if HAVE_TDM
    if (wave == 0 && kb + 1 < NKB)   // prefetch next K tile into other buffer
      tdm_load_1d((unsigned)(((kb+1) & 1) * KS_BYTES),
                  k + bbase + (size_t)(kb+1)*64*ND, KS_BYTES);
#else
    {
      bf16* Kd = (bf16*)(smem_raw + (kb & 1) * KS_BYTES);
      for (int i = 0; i < 16; ++i) {
        int cc = tid + i*256;
        *(bf16x8*)(Kd + cc*8) = *(const bf16x8*)(k + bbase + (size_t)kb*64*ND + cc*8);
      }
    }
#endif
    // V tile -> transposed LDS Vt[d][72]  (overlaps with in-flight TDM)
    for (int i = 0; i < 16; ++i) {
      int cc = tid + i*256;
      int kr = cc >> 6;
      int db = (cc & 63) * 8;
      bf16x8 vv = *(const bf16x8*)(v + bbase + (size_t)(kb*64+kr)*ND + db);
      #pragma unroll
      for (int j = 0; j < 8; ++j) Vt[(db+j)*72 + kr] = vv[j];
    }
#if HAVE_TDM
    if (wave == 0) {
      if (kb + 1 < NKB) __builtin_amdgcn_s_wait_tensorcnt(1);  // current K done, prefetch may fly
      else              __builtin_amdgcn_s_wait_tensorcnt(0);
    }
#endif
    __syncthreads();

    const bf16* Kcur = (bf16*)(smem_raw + (kb & 1) * KS_BYTES);
    const bf16* Kr[4];
    #pragma unroll
    for (int t = 0; t < 4; ++t) Kr[t] = Kcur + (size_t)(t*16 + ln16) * ND + offB;

    // scores: 16 queries x 64 keys, pipelined one d-step ahead
    f32x8 s[4];
    #pragma unroll
    for (int t = 0; t < 4; ++t) s[t] = fz;
    bf16x16 a = ldpair(QsRow, QsRow + 8);
    bf16x16 bm[4];
    #pragma unroll
    for (int t = 0; t < 4; ++t) bm[t] = ldpair(Kr[t], Kr[t] + 8);
    #pragma unroll
    for (int d0 = 0; d0 < ND; d0 += 32) {
      bf16x16 an = a;
      bf16x16 bn[4] = { bm[0], bm[1], bm[2], bm[3] };
      if (d0 + 32 < ND) {
        an = ldpair(QsRow + d0 + 32, QsRow + d0 + 40);
        #pragma unroll
        for (int t = 0; t < 4; ++t) bn[t] = ldpair(Kr[t] + d0 + 32, Kr[t] + d0 + 40);
      }
      #pragma unroll
      for (int t = 0; t < 4; ++t)
        s[t] = __builtin_amdgcn_wmma_f32_16x16x32_bf16(
                   false, a, false, bm[t], (short)0, s[t], false, false);
      a = an;
      #pragma unroll
      for (int t = 0; t < 4; ++t) bm[t] = bn[t];
    }

    // online softmax (rows r+8*half; reductions within each 16-lane half)
    float al[8];
    #pragma unroll
    for (int r = 0; r < 8; ++r) {
      #pragma unroll
      for (int t = 0; t < 4; ++t) s[t][r] *= sc2;
      float mx = fmaxf(fmaxf(s[0][r], s[1][r]), fmaxf(s[2][r], s[3][r]));
      mx = fmaxf(mx, __shfl_xor(mx, 1, 32));
      mx = fmaxf(mx, __shfl_xor(mx, 2, 32));
      mx = fmaxf(mx, __shfl_xor(mx, 4, 32));
      mx = fmaxf(mx, __shfl_xor(mx, 8, 32));
      float mnew = fmaxf(m_[r], mx);
      float aa = exp2f(m_[r] - mnew);
      float rs = 0.f;
      #pragma unroll
      for (int t = 0; t < 4; ++t) {
        float p = exp2f(s[t][r] - mnew);
        s[t][r] = p;
        rs += p;
      }
      rs += __shfl_xor(rs, 1, 32);
      rs += __shfl_xor(rs, 2, 32);
      rs += __shfl_xor(rs, 4, 32);
      rs += __shfl_xor(rs, 8, 32);
      l_[r] = aa * l_[r] + rs;
      m_[r] = mnew;
      al[r] = aa;
    }
    #pragma unroll
    for (int c = 0; c < 16; ++c)
      #pragma unroll
      for (int r = 0; r < 8; ++r) acc[c][r] *= al[r];

    // P (D-layout) -> per-wave LDS staging as bf16
    #pragma unroll
    for (int t = 0; t < 4; ++t)
      #pragma unroll
      for (int r = 0; r < 8; ++r)
        Pw[(r + 8*half) * 64 + t*16 + ln16] = (bf16)s[t][r];
    asm volatile("s_wait_dscnt 0" ::: "memory");

    // O += P @ V  (P in A-layout from staging, V^T rows as B), pipelined
    #pragma unroll
    for (int kk = 0; kk < 2; ++kk) {
      const bf16* pr = Pw + ln16*64 + kk*32 + offA;
      bf16x16 a2 = ldpair(pr, pr + 8);
      const bf16* vr0 = Vt + (size_t)(chalf*256 + ln16) * 72 + kk*32 + offB;
      bf16x16 bv = ldpair(vr0, vr0 + 8);
      #pragma unroll
      for (int c = 0; c < 16; ++c) {
        bf16x16 bnx = bv;
        if (c + 1 < 16) {
          const bf16* vr = vr0 + (size_t)(c+1)*16*72;
          bnx = ldpair(vr, vr + 8);
        }
        acc[c] = __builtin_amdgcn_wmma_f32_16x16x32_bf16(
                     false, a2, false, bv, (short)0, acc[c], false, false);
        bv = bnx;
      }
    }
  }

  // normalize + store fp32
  #pragma unroll
  for (int r = 0; r < 8; ++r) l_[r] = 1.f / l_[r];
  const int qrb = q0g + qsub*16;
  #pragma unroll
  for (int c = 0; c < 16; ++c) {
    int d = chalf*256 + c*16 + ln16;
    #pragma unroll
    for (int r = 0; r < 8; ++r) {
      int qr = qrb + r + 8*half;
      out[bbase + (size_t)qr * ND + d] = acc[c][r] * l_[r];
    }
  }
}

extern "C" void kernel_launch(void* const* d_in, const int* in_sizes, int n_in,
                              void* d_out, int out_size, void* d_ws, size_t ws_size,
                              hipStream_t stream) {
  const float* x  = (const float*)d_in[0];
  const float* Wq = (const float*)d_in[1];
  const float* bq = (const float*)d_in[2];
  const float* Wk = (const float*)d_in[3];
  const float* bk = (const float*)d_in[4];
  const float* Wv = (const float*)d_in[5];
  const float* bv = (const float*)d_in[6];
  float* out = (float*)d_out;

  const size_t NX = (size_t)NB * NS * ND;   // 8388608
  const size_t NW = (size_t)ND * ND;        // 262144

  bf16* xb  = (bf16*)d_ws;
  bf16* wqb = xb  + NX;
  bf16* wkb = wqb + NW;
  bf16* wvb = wkb + NW;
  bf16* qb  = wvb + NW;
  bf16* kb  = qb  + NX;
  bf16* vb  = kb  + NX;

  cvt_f32_bf16x8<<<(int)(NX/8/256), 256, 0, stream>>>((const f32x8*)x,  (bf16x8*)xb,  (int)(NX/8));
  cvt_f32_bf16x8<<<(int)(NW/8/256), 256, 0, stream>>>((const f32x8*)Wq, (bf16x8*)wqb, (int)(NW/8));
  cvt_f32_bf16x8<<<(int)(NW/8/256), 256, 0, stream>>>((const f32x8*)Wk, (bf16x8*)wkb, (int)(NW/8));
  cvt_f32_bf16x8<<<(int)(NW/8/256), 256, 0, stream>>>((const f32x8*)Wv, (bf16x8*)wvb, (int)(NW/8));

  qkv_gemm<<<dim3(128, 8, 3), 256, 0, stream>>>(xb, wqb, wkb, wvb, bq, bk, bv, qb, kb, vb);

  flash_attn<<<dim3(NS/64, NB), 256, SMEM_TOTAL, stream>>>(qb, kb, vb, out);
}
